// PushPullConv2DUnit_36120674959479
// MI455X (gfx1250) — matrix operand order repaired and checked by reading, not compiled, not run on hardware
//
#include <hip/hip_runtime.h>

typedef _Float16 half_t;
typedef __attribute__((ext_vector_type(16))) _Float16 v16h;
typedef __attribute__((ext_vector_type(8)))  _Float16 v8h;
typedef __attribute__((ext_vector_type(8)))  float    v8f;
typedef __attribute__((ext_vector_type(4)))  float    v4f;

#define BATCH 32
#define CIN   3
#define HIN   224
#define COUT  64
#define HOUT  112
#define KS    17      // 17x17 DoG surround

#define KQ    21      // kx*3 + c taps per ky row
#define KROW  32      // ky row padded to 32 -> one WMMA chunk per ky
#define KPAD  (7 * KROW)   // 224
#define XROW  704     // LDS x-tile row stride (halves): covers 6*111+31=697 reads

// ---- workspace layout (half_t elements) ----
// w16    : [64][224]  repacked f16 weights, k = ky*32 + (kx*3+c), zero padded
// push16 : [64][32][112][112] f16 ReLU'd push activations
#define W16_ELEMS (COUT * KPAD)
#define PUSH_OFF  W16_ELEMS

// ---------------- weights -> repacked padded f16 [64][224] ----------------
__global__ void cvt_w_kernel(const float* __restrict__ w, half_t* __restrict__ w16) {
    int i = blockIdx.x * 256 + threadIdx.x;
    if (i >= COUT * KPAD) return;
    int oc = i / KPAD;
    int k  = i - oc * KPAD;
    int ky = k >> 5;         // row
    int q  = k & 31;         // tap within row
    half_t v = (half_t)0.0f;
    if (q < KQ) {
        int kx = q / 3;
        int c  = q - 3 * kx;
        v = (half_t)w[oc * 147 + c * 49 + ky * 7 + kx];   // [64][3][7][7]
    }
    w16[i] = v;
}

// ---------------- stage 1: push conv (implicit GEMM, WMMA f16) ----------------
// block (224 thr = 7 waves) <-> (b, oy). wave <-> 16-wide x tile.
// A = im2col patches (M = x positions), B = weights (N = oc), D: lane stores
// 8 consecutive x of one oc as a single 16B store.
__global__ void __launch_bounds__(224)
push_conv_kernel(const float* __restrict__ x,
                 const half_t* __restrict__ w16,
                 half_t* __restrict__ push16) {
    __shared__ alignas(32) half_t xs[7 * XROW];      // staged input rows, ch-last, halo-padded
    __shared__ alignas(32) half_t wl[COUT * KPAD];   // weights copy

    int t   = threadIdx.x;
    int bid = blockIdx.x;           // 32*112
    int oy  = bid % HOUT;
    int b   = bid / HOUT;

    // prefetch input rows (emits global_prefetch_b8)
    if (t < 21) {
        int ky = t / 3, c = t - 3 * (t / 3);
        int iy = 2 * oy - 3 + ky;
        if (iy >= 0 && iy < HIN)
            __builtin_prefetch(&x[((size_t)(b * CIN + c) * HIN + iy) * HIN], 0, 1);
    }

    // copy weights to LDS with vector loads
    {
        const v8h* src = (const v8h*)w16;
        v8h* dst = (v8h*)wl;
        for (int i = t; i < (COUT * KPAD) / 8; i += 224) dst[i] = src[i];
    }

    // stage 7 input rows into LDS: index i = pix*3 + c, pix = ix+3 (halo), zeros elsewhere
    #pragma unroll
    for (int ky = 0; ky < 7; ++ky) {
        int  iy  = 2 * oy - 3 + ky;
        bool rok = (iy >= 0) && (iy < HIN);
        for (int i = t; i < XROW; i += 224) {
            half_t v = (half_t)0.0f;
            if (rok && i < 690) {
                int pix = i / 3;
                int c   = i - 3 * pix;
                int ix  = pix - 3;
                if (ix >= 0 && ix < HIN)
                    v = (half_t)x[((size_t)(b * CIN + c) * HIN + iy) * HIN + ix];
            }
            xs[ky * XROW + i] = v;
        }
    }
    __syncthreads();

    int lane = t & 31;
    int xt   = t >> 5;        // 0..6
    int l    = lane & 15;
    int hf   = lane >> 4;
    int ox   = xt * 16 + l;   // A-matrix row: this lane's output x

    // A taps: addr = ky*XROW + 6*ox + q ; A half h -> q = 16*hf + h
    const half_t* abase = &xs[6 * ox + 16 * hf];
    // B (weights): lane holds column oc = mt*16 + l, K = hf*16 + h contiguous
    const half_t* wbase = &wl[l * KPAD + hf * 16];

    v8f acc[4] = {{}, {}, {}, {}};
    #pragma unroll
    for (int ky = 0; ky < 7; ++ky) {
        v16h af;
        #pragma unroll
        for (int h = 0; h < 8; ++h) {
            af[h]     = abase[ky * XROW + h];
            af[h + 8] = abase[ky * XROW + 8 + h];
        }
        #pragma unroll
        for (int mt = 0; mt < 4; ++mt) {
            v16h bf = *(const v16h*)(wbase + mt * 16 * KPAD + ky * KROW); // 32B aligned
            acc[mt] = __builtin_amdgcn_wmma_f32_16x16x32_f16(false, af, false, bf,
                                                             (short)0, acc[mt], false, false);
        }
    }

    // store: lane holds x = xt*16 + 8*hf + r (r=0..7) for oc = mt*16 + l
    int xb0 = xt * 16 + 8 * hf;
    #pragma unroll
    for (int mt = 0; mt < 4; ++mt) {
        int oc = mt * 16 + l;
        v8h o;
        #pragma unroll
        for (int r = 0; r < 8; ++r) {
            float v = acc[mt][r];
            o[r] = (half_t)(v > 0.0f ? v : 0.0f);     // fused ReLU
        }
        *(v8h*)&push16[((size_t)(oc * BATCH + b) * HOUT + oy) * HOUT + xb0] = o;
    }
}

// ---------------- stage 2: depthwise 17x17 DoG via Toeplitz WMMA ----------------
// block (224 thr = 7 waves) <-> (c, b, yt); wave <-> x-tile. Shared zero-padded
// 32x128 LDS tile removes all edge branching; dy loop fully unrolled -> 17 WMMAs.
__global__ void __launch_bounds__(224)
surround_kernel(const half_t* __restrict__ push16,
                const float* __restrict__ bias,
                float* __restrict__ out) {
    __shared__ float gf[KS * KS];
    __shared__ float gsum;
    __shared__ alignas(32) half_t afr[KS * 512];     // [dy][lane][16] Toeplitz A frags
    __shared__ alignas(32) half_t tile[32 * 128];    // rows y0-8..y0+23, xi = x+8

    int t   = threadIdx.x;
    int bid = blockIdx.x;           // 64*32*7
    int yt  = bid % 7;
    int b   = (bid / 7) % BATCH;
    int c   = bid / (7 * BATCH);
    int y0  = yt * 16;

    // DoG kernel (deterministic serial normalization)
    for (int i = t; i < KS * KS; i += 224) {
        int yy = i / KS, xx = i - yy * KS;
        float fx = (float)(xx - 8), fy = (float)(yy - 8);
        float r2 = fx * fx + fy * fy;
        const float TWO_PI = 6.283185307179586f;
        float g1 = __expf(-r2 * (1.0f / 128.0f)) / (TWO_PI * 64.0f);  // sigma 8
        float g2 = __expf(-r2 * (1.0f / 8.0f))   / (TWO_PI * 4.0f);   // sigma 2
        float g = g1 - g2;
        gf[i] = g < 0.0f ? 0.0f : g;
    }
    __syncthreads();
    if (t == 0) {
        float s = 0.0f;
        for (int i = 0; i < KS * KS; ++i) s += gf[i];
        gsum = s;
    }
    __syncthreads();
    float inv = 1.0f / gsum;

    // Toeplitz A-fragments: A[m=x-off, j] = g[dy][j-m] for 0<=j-m<=16
    for (int i = t; i < KS * 512; i += 224) {
        int dy = i >> 9;
        int r  = i & 511;
        int ln = r >> 4;
        int h  = r & 15;
        int ll = ln & 15, hh = ln >> 4;
        int j  = hh * 16 + h;
        int d  = j - ll;
        float v = (d >= 0 && d <= 16) ? gf[dy * KS + d] * inv : 0.0f;
        afr[i] = (half_t)v;
    }

    // stage push tile (zero-padded: y in [y0-8, y0+24), x in [-8, 120))
    const half_t* plane = push16 + (size_t)(c * BATCH + b) * (HOUT * HOUT);
    for (int i = t; i < 32 * 128; i += 224) {
        int r  = i >> 7;
        int xi = i & 127;
        int y  = y0 - 8 + r;
        int x  = xi - 8;
        half_t v = (half_t)0.0f;
        if (y >= 0 && y < HOUT && x >= 0 && x < HOUT)
            v = plane[y * HOUT + x];
        tile[i] = v;
    }
    __syncthreads();

    int lane = t & 31;
    int xt   = t >> 5;
    int l    = lane & 15;
    int hf   = lane >> 4;
    int x0   = xt * 16;

    // B: lane = column n = l (y row), K = hf*16 + h -> contiguous halves, 32B aligned
    const half_t* tbase = &tile[l * 128 + x0 + hf * 16];
    const half_t* abase = &afr[lane * 16];

    v8f acc = {};
    #pragma unroll
    for (int dy = 0; dy < KS; ++dy) {
        v16h bf = *(const v16h*)(tbase + dy * 128);
        v16h af = *(const v16h*)(abase + dy * 512);
        acc = __builtin_amdgcn_wmma_f32_16x16x32_f16(false, af, false, bf,
                                                     (short)0, acc, false, false);
    }

    // epilogue: out = push - surr + bias  (push center re-read from LDS tile)
    float bc = bias[c];
    v8h pcv = *(const v8h*)&tile[(l + 8) * 128 + x0 + 8 * hf + 8];
    float res[8];
    #pragma unroll
    for (int r = 0; r < 8; ++r)
        res[r] = (float)pcv[r] - acc[r] + bc;      // M = r+8*hf -> x = x0+8*hf+r

    float* op = out + ((size_t)(b * COUT + c) * HOUT + (y0 + l)) * HOUT + x0 + 8 * hf;
    v4f o0 = { res[0], res[1], res[2], res[3] };
    v4f o1 = { res[4], res[5], res[6], res[7] };
    *(v4f*)op       = o0;
    *(v4f*)(op + 4) = o1;
}

extern "C" void kernel_launch(void* const* d_in, const int* in_sizes, int n_in,
                              void* d_out, int out_size, void* d_ws, size_t ws_size,
                              hipStream_t stream) {
    (void)in_sizes; (void)n_in; (void)out_size; (void)ws_size;

    const float* x      = (const float*)d_in[0];
    const float* push_w = (const float*)d_in[1];
    const float* bias   = (const float*)d_in[2];
    float*  out    = (float*)d_out;
    half_t* ws     = (half_t*)d_ws;
    half_t* w16    = ws;
    half_t* push16 = ws + PUSH_OFF;

    cvt_w_kernel<<<(COUT * KPAD + 255) / 256, 256, 0, stream>>>(push_w, w16);

    // stage 1: one block per (b, oy) output row
    push_conv_kernel<<<BATCH * HOUT, 224, 0, stream>>>(x, w16, push16);

    // stage 2: one block per (c, b, yt)
    surround_kernel<<<COUT * BATCH * 7, 224, 0, stream>>>(push16, bias, out);
}
